// CrossAttentionLayer_80101140070769
// MI455X (gfx1250) — matrix operand order repaired and checked by reading, not compiled, not run on hardware
//
#include <hip/hip_runtime.h>
#include <hip/hip_bf16.h>

typedef _Float16 v16h __attribute__((ext_vector_type(16)));
typedef _Float16 v8h  __attribute__((ext_vector_type(8)));
typedef float    v8f  __attribute__((ext_vector_type(8)));
typedef int      v4i  __attribute__((ext_vector_type(4)));

#define HIDDEN   2048
#define HEADS    16
#define HEAD_DIM 128
#define BATCH    2
#define SEQ      2048
#define MTOT     (BATCH * SEQ)
#define LN_EPS   1e-5f

__device__ __forceinline__ v8f wmma_f32_f16(v16h a, v16h b, v8f c) {
    return __builtin_amdgcn_wmma_f32_16x16x32_f16(false, a, false, b, (short)0, c, false, false);
}

// ---------------------------------------------------------------------------
// Async global->LDS copy (16B per lane). Builtin signature (probe-revealed):
//   (int4 AS1* gaddr, int4 AS3* lds, imm offset, imm cpol)
// Fallback: raw CDNA5 encoding per ISA 15.18.
// ---------------------------------------------------------------------------
#if __has_builtin(__builtin_amdgcn_global_load_async_to_lds_b128)
#define ASYNC_VIA_BUILTIN 1
#else
#define ASYNC_VIA_ASM 1
#endif

__device__ __forceinline__ void async_cp16(void* l, const void* g) {
#if defined(ASYNC_VIA_BUILTIN)
    __builtin_amdgcn_global_load_async_to_lds_b128(
        (__attribute__((address_space(1))) v4i*)g,
        (__attribute__((address_space(3))) v4i*)l, 0, 0);
#else
    unsigned loff = (unsigned)(size_t)l;
    asm volatile("global_load_async_to_lds_b128 %0, %1, off"
                 :: "v"(loff), "v"(g) : "memory");
#endif
}

template <int N>
__device__ __forceinline__ void wait_async() {
#if defined(ASYNC_VIA_BUILTIN) && __has_builtin(__builtin_amdgcn_s_wait_asynccnt)
    __builtin_amdgcn_s_wait_asynccnt(N);
#else
    asm volatile("s_wait_asynccnt %0" :: "n"(N) : "memory");
#endif
}

// ---------------------------------------------------------------------------
// Prep kernels: fp32 -> f16 convert (activations) and convert+transpose
// (weights, [K][N] f32 -> [N][K] f16) so GEMM B-fragments are contiguous.
// ---------------------------------------------------------------------------
__global__ __launch_bounds__(256) void cvt_f16_kernel(
    const float* __restrict__ src, _Float16* __restrict__ dst)
{
    size_t i = ((size_t)blockIdx.x * 256 + threadIdx.x) * 8;
    float4 a = *(const float4*)(src + i);
    float4 b = *(const float4*)(src + i + 4);
    v8h h = { (_Float16)a.x, (_Float16)a.y, (_Float16)a.z, (_Float16)a.w,
              (_Float16)b.x, (_Float16)b.y, (_Float16)b.z, (_Float16)b.w };
    *(v8h*)(dst + i) = h;
}

__global__ __launch_bounds__(256) void cvtT_f16_kernel(
    const float* __restrict__ W, _Float16* __restrict__ Wt)
{
    __shared__ _Float16 t[64][65];
    const int tid = threadIdx.x;
    const int k0 = blockIdx.y * 64, n0 = blockIdx.x * 64;
    #pragma unroll
    for (int j = 0; j < 16; ++j) {
        int idx = tid + 256 * j;
        int r = idx >> 6, c = idx & 63;
        t[c][r] = (_Float16)W[(size_t)(k0 + r) * HIDDEN + n0 + c];
    }
    __syncthreads();
    #pragma unroll
    for (int j = 0; j < 16; ++j) {
        int idx = tid + 256 * j;
        int r = idx >> 6, c = idx & 63;
        Wt[(size_t)(n0 + r) * HIDDEN + k0 + c] = t[r][c];
    }
}

// ---------------------------------------------------------------------------
// WMMA GEMM: C[M,N] = A[M,K](f16) * Bt[N,K](f16)^T + bias[N]
// 256 thr = 8 waves; tile 128x128, K-step 32, double-buffered async LDS.
// Wave w owns rows [w*16, w*16+16), all 8 N-fragments (8 WMMA / K-step).
// OUTMODE: 0 = f32, 1 = f16.
// ---------------------------------------------------------------------------
template <int OUTMODE>
__global__ __launch_bounds__(256) void gemm_bias_kernel(
    const _Float16* __restrict__ A, const _Float16* __restrict__ Bt,
    const float* __restrict__ bias, void* __restrict__ Cout,
    int M, int Kd, int N)
{
    __shared__ alignas(32) _Float16 smA[2][128 * 32];
    __shared__ alignas(32) _Float16 smB[2][128 * 32];
    const int tid  = threadIdx.x;
    const int w    = tid >> 5;
    const int lane = tid & 31;
    const int l15  = lane & 15;
    const int hi   = lane >> 4;
    const int m0   = blockIdx.y * 128;
    const int n0   = blockIdx.x * 128;
    const int moff = w * 16;

    const _Float16* Ap = A  + (size_t)m0 * Kd;
    const _Float16* Bp = Bt + (size_t)n0 * Kd;

    auto stage = [&](int buf, int k0) {   // 4 async instrs / thread / stage
        #pragma unroll
        for (int j = 0; j < 2; ++j) {
            int idx = tid + 256 * j;
            int row = idx >> 2;
            int kp  = (idx & 3) * 8;
            async_cp16(&smA[buf][row * 32 + kp], Ap + (size_t)row * Kd + k0 + kp);
            async_cp16(&smB[buf][row * 32 + kp], Bp + (size_t)row * Kd + k0 + kp);
        }
    };

    v8f acc[8] = {};
    const int NS = Kd / 32;
    stage(0, 0);
    for (int s = 0; s < NS; ++s) {
        const int buf = s & 1;
        if (s + 1 < NS) { stage(buf ^ 1, (s + 1) * 32); wait_async<4>(); }
        else            { wait_async<0>(); }
        __syncthreads();

        // A frag: lane=row(l15); elem k = hi*16+e
        int arow = moff + l15;
        v16h af = *(const v16h*)&smA[buf][arow * 32 + hi * 16];

        #pragma unroll
        for (int f = 0; f < 8; ++f) {
            // B frag: lane=col; elem k = hi*16+e, contiguous in Bt[n][k]
            v16h bf = *(const v16h*)&smB[buf][(f * 16 + l15) * 32 + hi * 16];
            acc[f] = wmma_f32_f16(af, bf, acc[f]);
        }
        __syncthreads();
    }

    #pragma unroll
    for (int f = 0; f < 8; ++f) {
        int col  = n0 + f * 16 + l15;
        float bv = bias[col];
        #pragma unroll
        for (int r = 0; r < 8; ++r) {
            int row = m0 + moff + hi * 8 + r;   // C: lane=col, VGPR r -> row r+hi*8
            float v = acc[f][r] + bv;
            if (OUTMODE == 1) ((_Float16*)Cout)[(size_t)row * N + col] = (_Float16)v;
            else              ((float*)Cout)[(size_t)row * N + col]    = v;
        }
    }
}

// ---------------------------------------------------------------------------
// Flash attention: block = (b,h) x 128 q rows; 8 waves x 16 rows.
// Q/K/V f16 [B,S,HIDDEN]; output f16 same layout. K tile staged async.
// ---------------------------------------------------------------------------
struct __align__(32) AttnSmem {
    _Float16 Ks[32 * 128];      // [key][d]
    _Float16 Vt[128 * 32];      // [d][key]
    _Float16 Pw[8 * 16 * 32];   // per-wave P tile
};

__global__ __launch_bounds__(256) void attention_kernel(
    const _Float16* __restrict__ Q, const _Float16* __restrict__ K,
    const _Float16* __restrict__ V, _Float16* __restrict__ Out)
{
    __shared__ AttnSmem sm;
    const int tid  = threadIdx.x;
    const int w    = tid >> 5;
    const int lane = tid & 31;
    const int l15  = lane & 15;
    const int hi   = lane >> 4;
    const int b    = blockIdx.y / HEADS;
    const int h    = blockIdx.y % HEADS;
    const int q0   = blockIdx.x * 128 + w * 16;
    const size_t base = (size_t)b * SEQ * HIDDEN + (size_t)h * HEAD_DIM;
    const float scale = 0.08838834764831845f;  // 1/sqrt(128)
    const _Float16* Kp = K + base;
    const _Float16* Vp = V + base;

    v16h qf[4];
    #pragma unroll
    for (int dc = 0; dc < 4; ++dc) {
        const _Float16* qp = Q + base + (size_t)(q0 + l15) * HIDDEN + dc * 32 + hi * 16;
        qf[dc] = *(const v16h*)qp;
    }

    float m_r[8], l_r[8];
    #pragma unroll
    for (int r = 0; r < 8; ++r) { m_r[r] = -3.0e38f; l_r[r] = 0.0f; }
    v8f o[8] = {};
    _Float16* Pw = &sm.Pw[w * 16 * 32];

    for (int kb = 0; kb < SEQ; kb += 32) {
        __syncthreads();
        // K tile: async DMA, no VGPR round-trip
        #pragma unroll
        for (int j = 0; j < 2; ++j) {
            int idx = tid + 256 * j;
            int key = idx >> 4;
            int dp  = (idx & 15) * 8;
            async_cp16(&sm.Ks[key * 128 + dp], Kp + (size_t)(kb + key) * HIDDEN + dp);
        }
        // V tile transposed into LDS [d][key]
        #pragma unroll
        for (int j = 0; j < 2; ++j) {
            int idx = tid + 256 * j;
            int key = idx >> 4;
            int dp  = (idx & 15) * 8;
            v8h vv = *(const v8h*)(Vp + (size_t)(kb + key) * HIDDEN + dp);
            #pragma unroll
            for (int i = 0; i < 8; ++i) sm.Vt[(dp + i) * 32 + key] = vv[i];
        }
        wait_async<0>();
        __syncthreads();

        v8f s0 = {}, s1 = {};
        #pragma unroll
        for (int dc = 0; dc < 4; ++dc) {
            v16h b0 = *(const v16h*)&sm.Ks[l15 * 128 + dc * 32 + hi * 16];
            v16h b1 = *(const v16h*)&sm.Ks[(16 + l15) * 128 + dc * 32 + hi * 16];
            s0 = wmma_f32_f16(qf[dc], b0, s0);
            s1 = wmma_f32_f16(qf[dc], b1, s1);
        }

        #pragma unroll
        for (int r = 0; r < 8; ++r) {
            float a0 = s0[r] * scale, a1 = s1[r] * scale;
            float mx = fmaxf(a0, a1);
            #pragma unroll
            for (int msk = 8; msk >= 1; msk >>= 1) mx = fmaxf(mx, __shfl_xor(mx, msk, 32));
            float mnew = fmaxf(m_r[r], mx);
            float corr = __expf(m_r[r] - mnew);
            float p0 = __expf(a0 - mnew), p1 = __expf(a1 - mnew);
            float rs = p0 + p1;
            #pragma unroll
            for (int msk = 8; msk >= 1; msk >>= 1) rs += __shfl_xor(rs, msk, 32);
            l_r[r] = l_r[r] * corr + rs;
            m_r[r] = mnew;
            #pragma unroll
            for (int f = 0; f < 8; ++f) o[f][r] *= corr;
            Pw[(r + hi * 8) * 32 + l15]      = (_Float16)p0;
            Pw[(r + hi * 8) * 32 + 16 + l15] = (_Float16)p1;
        }
        __builtin_amdgcn_wave_barrier();
        asm volatile("s_wait_dscnt 0" ::: "memory");

        v16h pf = *(const v16h*)&Pw[l15 * 32 + hi * 16];
        #pragma unroll
        for (int f = 0; f < 8; ++f) {
            v16h bv = *(const v16h*)&sm.Vt[(f * 16 + l15) * 32 + hi * 16];
            o[f] = wmma_f32_f16(pf, bv, o[f]);
        }
    }

    #pragma unroll
    for (int r = 0; r < 8; ++r) {
        float inv = 1.0f / l_r[r];
        #pragma unroll
        for (int f = 0; f < 8; ++f)
            Out[base + (size_t)(q0 + hi * 8 + r) * HIDDEN + f * 16 + l15] =
                (_Float16)(o[f][r] * inv);
    }
}

// ---------------------------------------------------------------------------
// out = LayerNorm(hidden + sigmoid(G) * proj)
// ---------------------------------------------------------------------------
__global__ __launch_bounds__(256) void gate_ln_kernel(
    const float* __restrict__ hidden, const _Float16* __restrict__ G,
    const float* __restrict__ proj, const float* __restrict__ gamma,
    const float* __restrict__ beta, float* __restrict__ out)
{
    const int row = blockIdx.x;
    const int tid = threadIdx.x;
    const size_t rb = (size_t)row * HIDDEN;

    float vals[8], s = 0.f, q = 0.f;
    #pragma unroll
    for (int i = 0; i < 8; ++i) {
        int j = tid + 256 * i;
        float g = 1.0f / (1.0f + __expf(-(float)G[rb + j]));
        float v = hidden[rb + j] + g * proj[rb + j];
        vals[i] = v; s += v; q += v * v;
    }
    #pragma unroll
    for (int msk = 16; msk >= 1; msk >>= 1) {
        s += __shfl_xor(s, msk, 32);
        q += __shfl_xor(q, msk, 32);
    }
    __shared__ float red[16];
    int wv = tid >> 5;
    if ((tid & 31) == 0) { red[wv] = s; red[8 + wv] = q; }
    __syncthreads();
    if (tid == 0) {
        float ts = 0.f, tq = 0.f;
        for (int i = 0; i < 8; ++i) { ts += red[i]; tq += red[8 + i]; }
        red[0] = ts; red[8] = tq;
    }
    __syncthreads();
    float mean = red[0] * (1.0f / HIDDEN);
    float var  = red[8] * (1.0f / HIDDEN) - mean * mean;
    float rstd = rsqrtf(var + LN_EPS);
    #pragma unroll
    for (int i = 0; i < 8; ++i) {
        int j = tid + 256 * i;
        out[rb + j] = (vals[i] - mean) * rstd * gamma[j] + beta[j];
    }
}

// ---------------------------------------------------------------------------
extern "C" void kernel_launch(void* const* d_in, const int* in_sizes, int n_in,
                              void* d_out, int out_size, void* d_ws, size_t ws_size,
                              hipStream_t stream)
{
    (void)in_sizes; (void)n_in; (void)out_size; (void)ws_size;
    const float* hidden = (const float*)d_in[0];
    const float* cross  = (const float*)d_in[1];
    const float* Wq = (const float*)d_in[2];  const float* bq = (const float*)d_in[3];
    const float* Wk = (const float*)d_in[4];  const float* bk = (const float*)d_in[5];
    const float* Wv = (const float*)d_in[6];  const float* bv = (const float*)d_in[7];
    const float* Wo = (const float*)d_in[8];  const float* bo = (const float*)d_in[9];
    const float* Wg = (const float*)d_in[10]; const float* bg = (const float*)d_in[11];
    const float* gamma = (const float*)d_in[12];
    const float* beta  = (const float*)d_in[13];
    float* out = (float*)d_out;

    char* ws = (char*)d_ws;
    const size_t E  = (size_t)MTOT * HIDDEN;           // activation elems
    const size_t E2 = E * 2;                           // f16 activation bytes
    const size_t W2 = (size_t)HIDDEN * HIDDEN * 2;     // f16 weight bytes
    _Float16* Ah  = (_Float16*)(ws);                   // dead after G gemm
    _Float16* WvT = (_Float16*)(ws + E2);              // dead after V gemm
    _Float16* WgT = (_Float16*)(ws + E2 + W2);         // dead after G gemm
    _Float16* Ch  = (_Float16*)(ws + E2 + 2 * W2);     // dead after V gemm
    _Float16* WqT = (_Float16*)(ws + 2 * E2 + 2 * W2);
    _Float16* WkT = (_Float16*)(ws + 2 * E2 + 3 * W2);
    _Float16* WoT = (_Float16*)(ws + 2 * E2 + 4 * W2);
    _Float16* Qf  = (_Float16*)(ws + 2 * E2 + 5 * W2);
    _Float16* Kf  = (_Float16*)(ws + 3 * E2 + 5 * W2);
    _Float16* Vf  = (_Float16*)(ws + 4 * E2 + 5 * W2);
    _Float16* Gh  = (_Float16*)(ws + 5 * E2 + 5 * W2);
    _Float16* At  = Ch;                                // overlay (Ch dead)
    float*    Pj  = (float*)(ws);                      // overlay Ah/WvT/WgT (dead)

    dim3 blk(256);
    // fp32 -> f16 activations; fp32 -> f16 transposed weights
    cvt_f16_kernel<<<dim3((unsigned)(E / 2048)), blk, 0, stream>>>(hidden, Ah);
    cvt_f16_kernel<<<dim3((unsigned)(E / 2048)), blk, 0, stream>>>(cross,  Ch);
    dim3 gt(HIDDEN / 64, HIDDEN / 64);
    cvtT_f16_kernel<<<gt, blk, 0, stream>>>(Wq, WqT);
    cvtT_f16_kernel<<<gt, blk, 0, stream>>>(Wk, WkT);
    cvtT_f16_kernel<<<gt, blk, 0, stream>>>(Wv, WvT);
    cvtT_f16_kernel<<<gt, blk, 0, stream>>>(Wg, WgT);
    cvtT_f16_kernel<<<gt, blk, 0, stream>>>(Wo, WoT);

    dim3 gg(HIDDEN / 128, MTOT / 128);
    gemm_bias_kernel<1><<<gg, blk, 0, stream>>>(Ah, WqT, bq, Qf, MTOT, HIDDEN, HIDDEN);
    gemm_bias_kernel<1><<<gg, blk, 0, stream>>>(Ch, WkT, bk, Kf, MTOT, HIDDEN, HIDDEN);
    gemm_bias_kernel<1><<<gg, blk, 0, stream>>>(Ch, WvT, bv, Vf, MTOT, HIDDEN, HIDDEN);
    gemm_bias_kernel<1><<<gg, blk, 0, stream>>>(Ah, WgT, bg, Gh, MTOT, HIDDEN, HIDDEN);

    dim3 ga(SEQ / 128, BATCH * HEADS);
    attention_kernel<<<ga, blk, 0, stream>>>(Qf, Kf, Vf, At);

    gemm_bias_kernel<0><<<gg, blk, 0, stream>>>(At, WoT, bo, Pj, MTOT, HIDDEN, HIDDEN);
    gate_ln_kernel<<<dim3(MTOT), blk, 0, stream>>>(hidden, Gh, Pj, gamma, beta, out);
}
